// GraphFilterRNNBatch_31679678775965
// MI455X (gfx1250) — matrix operand order repaired and checked by reading, not compiled, not run on hardware
//
#include <hip/hip_runtime.h>
#include <hip/hip_bf16.h>

// ---------------------------------------------------------------------------
// GraphFilterRNN batch kernel for MI455X (gfx1250), bf16 WMMA + f32 accumulate
// B=16, N=1024, F=G=H=128, E=1, K_TAPS=4
// Software-pipelined K-loops: loads for iteration i+1 issued before WMMAs of i.
// ---------------------------------------------------------------------------

#define NN   1024   // nodes
#define NB   16     // batches
#define KT   4      // filter taps

typedef __attribute__((ext_vector_type(16))) __bf16 v16bf;
typedef __attribute__((ext_vector_type(8)))  float  v8f;

union FragBF {          // 16 bf16 = 32 bytes = 8 VGPRs
    v16bf v;
    uint4 u[2];
};
union PairBF { __bf16 h[2]; unsigned u; };

__device__ __forceinline__ __bf16 f2bf(float f) { return (__bf16)f; }

// ---------------------------------------------------------------------------
// Pack weights: W_AB[g][k*256+f] = (f<128 ? wA : wB), W_D[g][k*128+f] = wD
// weight layout in memory: [H][E=1][K][F] row-major
// ---------------------------------------------------------------------------
__global__ void k_pack_weights(const float* __restrict__ wA,
                               const float* __restrict__ wB,
                               const float* __restrict__ wD,
                               __bf16* __restrict__ W_AB,
                               __bf16* __restrict__ W_D) {
    int idx = blockIdx.x * blockDim.x + threadIdx.x;
    if (idx < 128 * 1024) {
        int g = idx >> 10, kf = idx & 1023;
        int k = kf >> 8, f = kf & 255;
        float v = (f < 128) ? wA[(g * 4 + k) * 128 + f]
                            : wB[(g * 4 + k) * 128 + (f - 128)];
        W_AB[idx] = f2bf(v);
    } else {
        int i2 = idx - 128 * 1024;
        if (i2 < 128 * 512) {
            int g = i2 >> 9, kf = i2 & 511;
            int k = kf >> 7, f = kf & 127;
            W_D[i2] = f2bf(wD[(g * 4 + k) * 128 + f]);
        }
    }
}

// ---------------------------------------------------------------------------
// S^T: ST[b][n][m] = bf16(S[b][m][n]), LDS-tiled 32x32 transpose
// ---------------------------------------------------------------------------
__global__ void k_transpose_S(const float* __restrict__ S, __bf16* __restrict__ ST) {
    __shared__ float tile[32][33];
    int b = blockIdx.z;
    int n0 = blockIdx.x * 32;   // column base of S
    int m0 = blockIdx.y * 32;   // row base of S
    const float* Sb = S + (size_t)b * NN * NN;
    __bf16* STb = ST + (size_t)b * NN * NN;
    int tx = threadIdx.x, ty = threadIdx.y;
#pragma unroll
    for (int i = 0; i < 4; ++i)
        tile[ty + 8 * i][tx] = Sb[(size_t)(m0 + ty + 8 * i) * NN + n0 + tx];
    __syncthreads();
#pragma unroll
    for (int i = 0; i < 4; ++i)
        STb[(size_t)(n0 + ty + 8 * i) * NN + m0 + tx] = f2bf(tile[tx][ty + 8 * i]);
}

// ---------------------------------------------------------------------------
// Tap0: Zc[b][0][f][n] = bf16(concat(x,hidden)) and ZcT[b][n][f] (kf offset 0)
// ---------------------------------------------------------------------------
__global__ void k_tap0(const float* __restrict__ x, const float* __restrict__ hid,
                       __bf16* __restrict__ Zc, __bf16* __restrict__ ZcT) {
    __shared__ float tile[32][33];
    int b = blockIdx.z;
    int n0 = blockIdx.x * 32;
    int f0 = blockIdx.y * 32;   // 0..255 concat feature
    int tx = threadIdx.x, ty = threadIdx.y;
    __bf16* Zb  = Zc  + (size_t)b * (KT * 256 * NN);   // tap 0
    __bf16* ZTb = ZcT + (size_t)b * (NN * 1024);
#pragma unroll
    for (int i = 0; i < 4; ++i) {
        int f = f0 + ty + 8 * i;
        float v = (f < 128) ? x[((size_t)b * 128 + f) * NN + n0 + tx]
                            : hid[((size_t)b * 128 + (f - 128)) * NN + n0 + tx];
        Zb[(size_t)f * NN + n0 + tx] = f2bf(v);
        tile[ty + 8 * i][tx] = v;
    }
    __syncthreads();
#pragma unroll
    for (int i = 0; i < 4; ++i)
        ZTb[(size_t)(n0 + ty + 8 * i) * 1024 + f0 + tx] = f2bf(tile[tx][ty + 8 * i]);
}

// ---------------------------------------------------------------------------
// Diffusion: Z_k^T = S^T @ Z_{k-1}^T  (per batch)
//   A = S^T tile  (rows n_out, contiguous contraction m)   [n][m] bf16
//   B = Z_{k-1}   (cols f,     contiguous contraction m)   [f][n] bf16
// Writes Zout[f][n] (packed pairs) and ZT[n][koff+f].
// Wave = 16(M=n_out) x 64(4 N-tiles of features). Block = 8 waves = 128 M.
// ---------------------------------------------------------------------------
__global__ void __launch_bounds__(256)
k_diffuse(const __bf16* __restrict__ ST,
          const __bf16* __restrict__ Zin,
          __bf16* __restrict__ Zout,
          __bf16* __restrict__ ZT,
          int zbstride, int ztbstride, int ldT, int koff) {
    int b    = blockIdx.z;
    int wave = threadIdx.x >> 5;
    int lane = threadIdx.x & 31;
    int l16  = lane & 15;
    int hsel = lane >> 4;                    // lane half (0/1)
    int m0 = blockIdx.x * 128 + wave * 16;   // output node tile
    int f0 = blockIdx.y * 64;                // feature group (4 tiles)

    const __bf16* STb = ST + (size_t)b * NN * NN;
    const __bf16* Zb  = Zin  + (size_t)b * zbstride;
    __bf16* Zob = Zout + (size_t)b * zbstride;
    __bf16* ZTb = ZT   + (size_t)b * ztbstride;

    v8f acc[4] = {};
    // A: lanes 0-15 need K {0..7,16..23}; lanes 16-31 need K {8..15,24..31}
    const __bf16* aPtr  = STb + (size_t)(m0 + l16) * NN + 8 * hsel;
    // B: lanes 0-15 hold K {0..7,16..23}; lanes 16-31 hold K {8..15,24..31}
    const __bf16* bPtr0 = Zb  + (size_t)(f0 + l16) * NN + 8 * hsel;

    FragBF a[2];
    FragBF bb[2][4];
    // prologue: buffer 0 <- kk = 0
    a[0].u[0] = *(const uint4*)(aPtr);
    a[0].u[1] = *(const uint4*)(aPtr + 16);
#pragma unroll
    for (int t = 0; t < 4; ++t) {
        const __bf16* bp = bPtr0 + (size_t)t * 16 * NN;
        bb[0][t].u[0] = *(const uint4*)(bp);
        bb[0][t].u[1] = *(const uint4*)(bp + 16);
    }
#pragma unroll
    for (int i = 0; i < NN / 32; ++i) {
        const int cur = i & 1;
        const int nxt = cur ^ 1;
        const int kn = (i + 1) * 32;
        if (i + 1 < NN / 32) {               // issue next iteration's loads first
            a[nxt].u[0] = *(const uint4*)(aPtr + kn);
            a[nxt].u[1] = *(const uint4*)(aPtr + kn + 16);
            __builtin_prefetch(aPtr + kn + 224, 0, 1);   // global_prefetch_b8
#pragma unroll
            for (int t = 0; t < 4; ++t) {
                const __bf16* bp = bPtr0 + (size_t)t * 16 * NN + kn;
                bb[nxt][t].u[0] = *(const uint4*)(bp);
                bb[nxt][t].u[1] = *(const uint4*)(bp + 16);
            }
        }
#pragma unroll
        for (int t = 0; t < 4; ++t)
            acc[t] = __builtin_amdgcn_wmma_f32_16x16x32_bf16(
                false, a[cur].v, false, bb[cur][t].v, (short)0, acc[t], false, false);
    }

    int mbase = m0 + 8 * hsel;               // D: M = r + 8*hsel, N = l16
#pragma unroll
    for (int t = 0; t < 4; ++t) {
        int f = f0 + t * 16 + l16;
        // normal layout [f][n], n contiguous across r -> packed pair stores
#pragma unroll
        for (int r = 0; r < 8; r += 2) {
            PairBF p;
            p.h[0] = f2bf(acc[t][r]);
            p.h[1] = f2bf(acc[t][r + 1]);
            *(unsigned*)(Zob + (size_t)f * NN + mbase + r) = p.u;
        }
        // transposed layout [n][koff+f]
#pragma unroll
        for (int r = 0; r < 8; ++r)
            ZTb[(size_t)(mbase + r) * ldT + koff + f] = f2bf(acc[t][r]);
    }
}

// ---------------------------------------------------------------------------
// h_next = relu(W_AB @ Zc_all + biasA + biasB)  -> Y tap0 ([f][n] and [n][f])
//   A = W_AB [128][1024] bf16, B = ZcT[b][n][kf] bf16, K=1024
// ---------------------------------------------------------------------------
__global__ void __launch_bounds__(256)
k_wgemm_hnext(const __bf16* __restrict__ W,
              const __bf16* __restrict__ ZcT,
              const float* __restrict__ biasA, const float* __restrict__ biasB,
              __bf16* __restrict__ Ynorm, __bf16* __restrict__ YT,
              int zt_bstride, int y_bstride, int yt_bstride) {
    int b    = blockIdx.z;
    int wave = threadIdx.x >> 5;
    int lane = threadIdx.x & 31;
    int l16  = lane & 15;
    int hsel = lane >> 4;
    int g0 = wave * 16;                      // 8 waves cover all 128 rows
    int n0 = blockIdx.x * 64;

    const __bf16* ZTb = ZcT + (size_t)b * zt_bstride;
    v8f acc[4] = {};
    const __bf16* aPtr  = W   + (size_t)(g0 + l16) * 1024 + 8 * hsel;
    const __bf16* bPtr0 = ZTb + (size_t)(n0 + l16) * 1024 + 8 * hsel;

    FragBF a[2];
    FragBF bb[2][4];
    a[0].u[0] = *(const uint4*)(aPtr);
    a[0].u[1] = *(const uint4*)(aPtr + 16);
#pragma unroll
    for (int t = 0; t < 4; ++t) {
        const __bf16* bp = bPtr0 + (size_t)t * 16 * 1024;
        bb[0][t].u[0] = *(const uint4*)(bp);
        bb[0][t].u[1] = *(const uint4*)(bp + 16);
    }
#pragma unroll
    for (int i = 0; i < 1024 / 32; ++i) {
        const int cur = i & 1;
        const int nxt = cur ^ 1;
        const int kn = (i + 1) * 32;
        if (i + 1 < 1024 / 32) {
            a[nxt].u[0] = *(const uint4*)(aPtr + kn);
            a[nxt].u[1] = *(const uint4*)(aPtr + kn + 16);
#pragma unroll
            for (int t = 0; t < 4; ++t) {
                const __bf16* bp = bPtr0 + (size_t)t * 16 * 1024 + kn;
                bb[nxt][t].u[0] = *(const uint4*)(bp);
                bb[nxt][t].u[1] = *(const uint4*)(bp + 16);
            }
        }
#pragma unroll
        for (int t = 0; t < 4; ++t)
            acc[t] = __builtin_amdgcn_wmma_f32_16x16x32_bf16(
                false, a[cur].v, false, bb[cur][t].v, (short)0, acc[t], false, false);
    }

    __bf16* Yb  = Ynorm + (size_t)b * y_bstride;
    __bf16* YTb = YT    + (size_t)b * yt_bstride;
    int gb = g0 + 8 * hsel;
#pragma unroll
    for (int t = 0; t < 4; ++t) {
        int n = n0 + t * 16 + l16;
        float v[8];
#pragma unroll
        for (int r = 0; r < 8; ++r) {
            int g = gb + r;
            float u = acc[t][r] + biasA[g] + biasB[g];
            v[r] = fmaxf(u, 0.0f);
            Yb[(size_t)g * NN + n] = f2bf(v[r]);   // [f][n] for diffusion B
        }
#pragma unroll
        for (int r = 0; r < 8; r += 2) {           // YT[n][g], kf offset 0
            PairBF p;
            p.h[0] = f2bf(v[r]);
            p.h[1] = f2bf(v[r + 1]);
            *(unsigned*)(YTb + (size_t)n * 512 + gb + r) = p.u;
        }
    }
}

// ---------------------------------------------------------------------------
// u = W_D @ Y_all + biasD  -> d_out fp32
//   A = W_D [128][512] bf16, B = YT[b][n][kf] bf16, K=512
// ---------------------------------------------------------------------------
__global__ void __launch_bounds__(256)
k_wgemm_out(const __bf16* __restrict__ W,
            const __bf16* __restrict__ YT,
            const float* __restrict__ biasD,
            float* __restrict__ out, int yt_bstride) {
    int b    = blockIdx.z;
    int wave = threadIdx.x >> 5;
    int lane = threadIdx.x & 31;
    int l16  = lane & 15;
    int hsel = lane >> 4;
    int g0 = wave * 16;
    int n0 = blockIdx.x * 64;

    const __bf16* YTb = YT + (size_t)b * yt_bstride;
    v8f acc[4] = {};
    const __bf16* aPtr  = W   + (size_t)(g0 + l16) * 512 + 8 * hsel;
    const __bf16* bPtr0 = YTb + (size_t)(n0 + l16) * 512 + 8 * hsel;

    FragBF a[2];
    FragBF bb[2][4];
    a[0].u[0] = *(const uint4*)(aPtr);
    a[0].u[1] = *(const uint4*)(aPtr + 16);
#pragma unroll
    for (int t = 0; t < 4; ++t) {
        const __bf16* bp = bPtr0 + (size_t)t * 16 * 512;
        bb[0][t].u[0] = *(const uint4*)(bp);
        bb[0][t].u[1] = *(const uint4*)(bp + 16);
    }
#pragma unroll
    for (int i = 0; i < 512 / 32; ++i) {
        const int cur = i & 1;
        const int nxt = cur ^ 1;
        const int kn = (i + 1) * 32;
        if (i + 1 < 512 / 32) {
            a[nxt].u[0] = *(const uint4*)(aPtr + kn);
            a[nxt].u[1] = *(const uint4*)(aPtr + kn + 16);
#pragma unroll
            for (int t = 0; t < 4; ++t) {
                const __bf16* bp = bPtr0 + (size_t)t * 16 * 512 + kn;
                bb[nxt][t].u[0] = *(const uint4*)(bp);
                bb[nxt][t].u[1] = *(const uint4*)(bp + 16);
            }
        }
#pragma unroll
        for (int t = 0; t < 4; ++t)
            acc[t] = __builtin_amdgcn_wmma_f32_16x16x32_bf16(
                false, a[cur].v, false, bb[cur][t].v, (short)0, acc[t], false, false);
    }

    float* ob = out + (size_t)b * 128 * NN;
    int gb = g0 + 8 * hsel;
#pragma unroll
    for (int t = 0; t < 4; ++t) {
        int n = n0 + t * 16 + l16;
#pragma unroll
        for (int r = 0; r < 8; ++r) {
            int g = gb + r;
            ob[(size_t)g * NN + n] = acc[t][r] + biasD[g];
        }
    }
}

// ---------------------------------------------------------------------------
// Launch
// ---------------------------------------------------------------------------
extern "C" void kernel_launch(void* const* d_in, const int* in_sizes, int n_in,
                              void* d_out, int out_size, void* d_ws, size_t ws_size,
                              hipStream_t stream) {
    const float* x   = (const float*)d_in[0];
    const float* hid = (const float*)d_in[1];
    const float* S   = (const float*)d_in[2];
    const float* wA  = (const float*)d_in[3];
    const float* wB  = (const float*)d_in[4];
    const float* wD  = (const float*)d_in[5];
    const float* bA  = (const float*)d_in[6];
    const float* bB  = (const float*)d_in[7];
    const float* bD  = (const float*)d_in[8];
    float* out = (float*)d_out;

    char* ws = (char*)d_ws;
    __bf16* ST   = (__bf16*)(ws);                            // 32 MB  [16][1024][1024]
    __bf16* Zc   = (__bf16*)(ws + 33554432);                 // 32 MB  [16][4][256][1024]
    __bf16* ZcT  = (__bf16*)(ws + 67108864);                 // 32 MB  [16][1024][1024]
    __bf16* W_AB = (__bf16*)(ws + 100663296);                // 256 KB [128][1024]
    __bf16* W_D  = (__bf16*)(ws + 100925440);                // 128 KB [128][512]
    // Reuse the dead Zc region after the h_next GEMM:
    __bf16* Ynorm = (__bf16*)(ws + 33554432);                // 16 MB [16][4][128][1024]
    __bf16* YT    = (__bf16*)(ws + 33554432 + 16777216);     // 16 MB [16][1024][512]

    // 1) pack weights to bf16
    k_pack_weights<<<dim3((128 * 1024 + 128 * 512) / 256), dim3(256), 0, stream>>>(
        wA, wB, wD, W_AB, W_D);

    // 2) S^T bf16
    k_transpose_S<<<dim3(32, 32, NB), dim3(32, 8), 0, stream>>>(S, ST);

    // 3) tap0 of concat(x, hidden) in both layouts
    k_tap0<<<dim3(32, 8, NB), dim3(32, 8), 0, stream>>>(x, hid, Zc, ZcT);

    // 4) diffusion of concat taps (F=256): Zc[k] = Zc[k-1] @ S
    for (int k = 1; k < KT; ++k) {
        k_diffuse<<<dim3(8, 4, NB), dim3(256), 0, stream>>>(
            ST, Zc + (size_t)(k - 1) * 256 * NN, Zc + (size_t)k * 256 * NN, ZcT,
            /*zbstride*/ KT * 256 * NN, /*ztbstride*/ NN * 1024,
            /*ldT*/ 1024, /*koff*/ k * 256);
    }

    // 5) h_next = relu(W_AB @ Zc_all + bA + bB)  (writes Y tap0, both layouts)
    k_wgemm_hnext<<<dim3(16, 1, NB), dim3(256), 0, stream>>>(
        W_AB, ZcT, bA, bB, Ynorm, YT,
        /*zt_bstride*/ NN * 1024, /*y_bstride*/ KT * 128 * NN, /*yt_bstride*/ NN * 512);

    // 6) diffusion of h_next taps (F=128)
    for (int k = 1; k < KT; ++k) {
        k_diffuse<<<dim3(8, 2, NB), dim3(256), 0, stream>>>(
            ST, Ynorm + (size_t)(k - 1) * 128 * NN, Ynorm + (size_t)k * 128 * NN, YT,
            /*zbstride*/ KT * 128 * NN, /*ztbstride*/ NN * 512,
            /*ldT*/ 512, /*koff*/ k * 128);
    }

    // 7) u = W_D @ Y_all + bD  -> fp32 output
    k_wgemm_out<<<dim3(16, 1, NB), dim3(256), 0, stream>>>(
        W_D, YT, bD, out, /*yt_bstride*/ NN * 512);
}
